// Bert_2963527435009
// MI455X (gfx1250) — matrix operand order repaired and checked by reading, not compiled, not run on hardware
//
#include <hip/hip_runtime.h>

// ---------------------------------------------------------------------------
// BERT-base forward for MI455X (gfx1250).
// All GEMM operands are bf16 in memory (weights pre-converted once, activations
// converted in epilogues), f32 accumulate via v_wmma_f32_16x16x32_bf16.
// LDS staging is a pure copy -> async global->LDS (ASYNCcnt path).
// ---------------------------------------------------------------------------

typedef __bf16 bf16_t;
typedef bf16_t v16bf __attribute__((ext_vector_type(16)));
typedef bf16_t v8bf  __attribute__((ext_vector_type(8)));
typedef float  v8f   __attribute__((ext_vector_type(8)));
typedef int    v4i_vs __attribute__((vector_size(16)));   // matches builtin param

union FragBF16 {
    v16bf v;
    v8bf  h[2];
};

__device__ __forceinline__ unsigned short f2bf(float f) {
    union { float f; unsigned int u; } cv;
    cv.f = f;
    unsigned int u = cv.u;
    unsigned int r = (u + 0x7FFFu + ((u >> 16) & 1u)) >> 16;  // RNE
    return (unsigned short)r;
}

__device__ __forceinline__ float bf2f(unsigned short b) {
    union { float f; unsigned int u; } cv;
    cv.u = ((unsigned int)b) << 16;
    return cv.f;
}

// 16-byte global->LDS copy; async path on gfx1250 toolchains that expose it.
__device__ __forceinline__ void cp16_g2l(unsigned short* lds, const unsigned short* g) {
#if __has_builtin(__builtin_amdgcn_global_load_async_to_lds_b128)
    __builtin_amdgcn_global_load_async_to_lds_b128(
        (__attribute__((address_space(1))) v4i_vs*)g,
        (__attribute__((address_space(3))) v4i_vs*)lds, 0, 0);
#else
    *(uint4*)lds = *(const uint4*)g;
#endif
}

__device__ __forceinline__ void wait_g2l() {
#if __has_builtin(__builtin_amdgcn_global_load_async_to_lds_b128)
#if __has_builtin(__builtin_amdgcn_s_wait_asynccnt)
    __builtin_amdgcn_s_wait_asynccnt(0);
#else
    asm volatile("s_wait_asynccnt 0x0" ::: "memory");
#endif
#endif
}

// ---------------------------------------------------------------------------
// Tiled bf16 GEMM:  acc = A[M,K] * B[N,K]^T  (both bf16 row-major)
//   epilogue: cv = acc*alpha + bias[n] (+resid f32) ; mask==0 -> -1e9
//   outMode 0: f32 store  (C = float*)
//   outMode 1: bf16 store, same [m][n] layout        (C = ushort*)
//   outMode 2: bf16 store, head-transposed:          (C = ushort*)
//              gm=b*S2+s, gn=h*DK2+dh  ->  C[(((b*H2+h)*DK2+dh)*S2)+s]
//   Batched over blockIdx.z = zb*Hh + zh.
//   BM=BN=128, BK=32; 256 threads = 8 waves; wave tile 32x64 (2x4 accs).
// ---------------------------------------------------------------------------
__global__ __launch_bounds__(256)
void gemm_bf16_wmma(const unsigned short* __restrict__ A,
                    const unsigned short* __restrict__ Bm,
                    const float* __restrict__ bias, const float* __restrict__ resid,
                    const float* __restrict__ mask, void* __restrict__ Cout,
                    int M, int N, int K, int lda, int ldb, int ldc,
                    long long sAb, long long sAh, long long sBb, long long sBh,
                    long long sCb, long long sCh,
                    int Hh, float alpha, int outMode, int maskS,
                    int S2, int H2, int DK2)
{
    const int tid = threadIdx.x;
    const int m0  = blockIdx.y * 128;
    const int n0  = blockIdx.x * 128;
    const int zb  = (int)blockIdx.z / Hh;
    const int zh  = (int)blockIdx.z % Hh;

    A  += (long long)zb * sAb + (long long)zh * sAh;
    Bm += (long long)zb * sBb + (long long)zh * sBh;

    __shared__ __align__(16) unsigned short As[128][40];   // [m][k], +8 pad
    __shared__ __align__(16) unsigned short Bs[128][40];   // [n][k], +8 pad

    const v8f vzero = {0.f, 0.f, 0.f, 0.f, 0.f, 0.f, 0.f, 0.f};
    v8f acc[2][4];
#pragma unroll
    for (int i = 0; i < 2; ++i)
#pragma unroll
        for (int j = 0; j < 4; ++j) acc[i][j] = vzero;

    const int lane = tid & 31;
    const int wave = tid >> 5;
    const int wm   = (wave >> 1) * 32;
    const int wn   = (wave & 1) * 64;
    const int l15  = lane & 15;
    const int aOff = (lane < 16) ? 0 : 8;
    const int bOff = (lane < 16) ? 0 : 8;

    // staging coordinates (pure copy: 32 bytes = 2 x b128 per thread per tile)
    const int cr  = tid >> 1;            // 0..127
    const int cc  = (tid & 1) << 4;      // 0 or 16 (bf16 elements)
    int brow = n0 + cr;                  // clamp for N<128 tail (P*V, N=64)
    if (brow >= N) brow = N - 1;

    const int nKT = K >> 5;
    for (int kt = 0; kt < nKT; ++kt) {
        {
            const unsigned short* ga = A + (long long)(m0 + cr) * lda
                                         + (long long)kt * 32 + cc;
            cp16_g2l(&As[cr][cc], ga);
            cp16_g2l(&As[cr][cc + 8], ga + 8);
            const unsigned short* gb = Bm + (long long)brow * ldb
                                          + (long long)kt * 32 + cc;
            cp16_g2l(&Bs[cr][cc], gb);
            cp16_g2l(&Bs[cr][cc + 8], gb + 8);
        }
        wait_g2l();
        __syncthreads();

        FragBF16 fa[2], fb[4];
#pragma unroll
        for (int t = 0; t < 2; ++t) {
            const unsigned short* p = &As[wm + t * 16 + l15][0];
            fa[t].h[0] = *(const v8bf*)(p + aOff);
            fa[t].h[1] = *(const v8bf*)(p + 16 + aOff);
        }
#pragma unroll
        for (int t = 0; t < 4; ++t) {
            const unsigned short* p = &Bs[wn + t * 16 + l15][0];
            fb[t].h[0] = *(const v8bf*)(p + bOff);
            fb[t].h[1] = *(const v8bf*)(p + 16 + bOff);
        }
#pragma unroll
        for (int tm = 0; tm < 2; ++tm)
#pragma unroll
            for (int tn = 0; tn < 4; ++tn)
                acc[tm][tn] = __builtin_amdgcn_wmma_f32_16x16x32_bf16(
                    false, fa[tm].v, false, fb[tn].v,
                    (short)0, acc[tm][tn], false, false);
        __syncthreads();
    }

    // ---- epilogue ------------------------------------------------------
    const int rowHalf = (lane < 16) ? 0 : 8;
    float*          Cf = (float*)Cout;
    unsigned short* Cb = (unsigned short*)Cout;
    if (outMode != 2) {
        Cf += (long long)zb * sCb + (long long)zh * sCh;
        Cb += (long long)zb * sCb + (long long)zh * sCh;
    }

#pragma unroll
    for (int tm = 0; tm < 2; ++tm) {
#pragma unroll
        for (int tn = 0; tn < 4; ++tn) {
            const int gn = n0 + wn + tn * 16 + l15;
            if (gn >= N) continue;
            const float bv = bias ? bias[gn] : 0.f;
            const bool masked = (mask != nullptr) &&
                                (mask[(long long)zb * maskS + gn] == 0.f);
#pragma unroll
            for (int r = 0; r < 8; ++r) {
                const int gm = m0 + wm + tm * 16 + rowHalf + r;
                float cv = acc[tm][tn][r] * alpha + bv;
                if (resid) cv += resid[(long long)gm * ldc + gn];
                if (masked) cv = -1e9f;
                if (outMode == 0) {
                    Cf[(long long)gm * ldc + gn] = cv;
                } else if (outMode == 1) {
                    Cb[(long long)gm * ldc + gn] = f2bf(cv);
                } else {
                    const int b  = gm / S2, s = gm % S2;
                    const int hh = gn / DK2, dh = gn % DK2;
                    Cb[(((long long)b * H2 + hh) * DK2 + dh) * S2 + s] = f2bf(cv);
                }
            }
        }
    }
}

// ---------------------------------------------------------------------------
// Row softmax, in place on bf16 rows of length S.
// ---------------------------------------------------------------------------
__global__ __launch_bounds__(256)
void softmax_rows_bf16(unsigned short* __restrict__ p, int S)
{
    unsigned short* row = p + (long long)blockIdx.x * S;
    const int tid = threadIdx.x;
    __shared__ float red[256];

    float m = -1e30f;
    for (int i = tid; i < S; i += 256) m = fmaxf(m, bf2f(row[i]));
    red[tid] = m;
    __syncthreads();
    for (int s = 128; s > 0; s >>= 1) {
        if (tid < s) red[tid] = fmaxf(red[tid], red[tid + s]);
        __syncthreads();
    }
    m = red[0];
    __syncthreads();

    float sum = 0.f;
    float ev[8];                                 // S/256 <= 2 used; headroom
    int   nE = 0;
    for (int i = tid; i < S; i += 256) {
        const float e = __expf(bf2f(row[i]) - m);
        ev[nE++] = e;
        sum += e;
    }
    red[tid] = sum;
    __syncthreads();
    for (int s = 128; s > 0; s >>= 1) {
        if (tid < s) red[tid] += red[tid + s];
        __syncthreads();
    }
    const float inv = 1.f / red[0];
    nE = 0;
    for (int i = tid; i < S; i += 256) row[i] = f2bf(ev[nE++] * inv);
}

// ---------------------------------------------------------------------------
// Row LayerNorm with optional f32 and/or bf16 outputs.
// ---------------------------------------------------------------------------
__global__ __launch_bounds__(256)
void layernorm_rows(const float* __restrict__ in, const float* __restrict__ g,
                    const float* __restrict__ b, float* __restrict__ outF,
                    unsigned short* __restrict__ outB, int D, float eps)
{
    const float* row = in + (long long)blockIdx.x * D;
    const int tid = threadIdx.x;
    __shared__ float s1[256];
    __shared__ float s2[256];

    float a = 0.f, a2 = 0.f;
    for (int i = tid; i < D; i += 256) {
        const float v = row[i];
        a += v; a2 += v * v;
    }
    s1[tid] = a; s2[tid] = a2;
    __syncthreads();
    for (int s = 128; s > 0; s >>= 1) {
        if (tid < s) { s1[tid] += s1[tid + s]; s2[tid] += s2[tid + s]; }
        __syncthreads();
    }
    const float mean = s1[0] / (float)D;
    const float var  = s2[0] / (float)D - mean * mean;
    const float rstd = rsqrtf(var + eps);
    __syncthreads();
    for (int i = tid; i < D; i += 256) {
        const float y = (row[i] - mean) * rstd * g[i] + b[i];
        if (outF) outF[(long long)blockIdx.x * D + i] = y;
        if (outB) outB[(long long)blockIdx.x * D + i] = f2bf(y);
    }
}

// ---------------------------------------------------------------------------
// Embedding gather (f32 raw; LN follows).
// ---------------------------------------------------------------------------
__global__ __launch_bounds__(256)
void embed_kernel(const int* __restrict__ ids, const int* __restrict__ tts,
                  const float* __restrict__ we, const float* __restrict__ pe,
                  const float* __restrict__ te, float* __restrict__ x,
                  int S, int D)
{
    const int t  = blockIdx.x;
    const int s  = t % S;
    const int id = ids[t];
    const int tt = tts[t];
    const float* wrow = we + (long long)id * D;
    const float* prow = pe + (long long)s * D;
    const float* trow = te + (long long)tt * D;
    float* xrow = x + (long long)t * D;
    for (int i = threadIdx.x; i < D; i += 256)
        xrow[i] = wrow[i] + trow[i] + prow[i];
}

// ---------------------------------------------------------------------------
// One-time f32 -> bf16 conversion (weights).
// ---------------------------------------------------------------------------
__global__ __launch_bounds__(256)
void convert_f32_bf16(const float* __restrict__ in, unsigned short* __restrict__ out,
                      long long n)
{
    const long long stride = (long long)gridDim.x * 256 * 4;
    for (long long i = ((long long)blockIdx.x * 256 + threadIdx.x) * 4; i < n; i += stride) {
        const float4 f = *(const float4*)(in + i);
        out[i + 0] = f2bf(f.x);
        out[i + 1] = f2bf(f.y);
        out[i + 2] = f2bf(f.z);
        out[i + 3] = f2bf(f.w);
    }
}

// ---------------------------------------------------------------------------
// Host orchestration
// ---------------------------------------------------------------------------
extern "C" void kernel_launch(void* const* d_in, const int* in_sizes, int n_in,
                              void* d_out, int out_size, void* d_ws, size_t ws_size,
                              hipStream_t stream)
{
    (void)in_sizes; (void)n_in; (void)out_size; (void)ws_size;

    const int*   input_ids  = (const int*)  d_in[0];
    const int*   token_type = (const int*)  d_in[1];
    const float* attn_mask  = (const float*)d_in[2];
    const float* word_emb   = (const float*)d_in[3];
    const float* pos_emb    = (const float*)d_in[4];
    const float* type_emb   = (const float*)d_in[5];
    const float* emb_g      = (const float*)d_in[6];
    const float* emb_b      = (const float*)d_in[7];
    const float* Wq = (const float*)d_in[8];
    const float* bq = (const float*)d_in[9];
    const float* Wk = (const float*)d_in[10];
    const float* bk = (const float*)d_in[11];
    const float* Wv = (const float*)d_in[12];
    const float* bv = (const float*)d_in[13];
    const float* Wo = (const float*)d_in[14];
    const float* bo = (const float*)d_in[15];
    const float* ag = (const float*)d_in[16];
    const float* ab = (const float*)d_in[17];
    const float* W1 = (const float*)d_in[18];
    const float* b1 = (const float*)d_in[19];
    const float* W2 = (const float*)d_in[20];
    const float* b2 = (const float*)d_in[21];
    const float* fg = (const float*)d_in[22];
    const float* fb = (const float*)d_in[23];

    constexpr int Bb = 8, S = 512, Dd = 768, Hh = 12, Ff = 3072, Ll = 12, DK = 64;
    constexpr int Mtok = Bb * S;                         // 4096
    constexpr long long MD  = (long long)Mtok * Dd;      // 3.15M
    constexpr long long MF  = (long long)Mtok * Ff;      // 12.6M
    constexpr long long SC  = (long long)Bb * Hh * S * S;// 25.2M
    constexpr long long LDD = (long long)Ll * Dd * Dd;
    constexpr long long LFD = (long long)Ll * Ff * Dd;
    const float SCALE = 0.125f;

    float* x = (float*)d_ws;                 // layer input (f32, residual use)
    float* t = x + MD;                       // pre-LN temporaries
    unsigned short* xbf = (unsigned short*)(t + MD);
    unsigned short* qbf = xbf + MD;          // Q bf16 / later ctx bf16
    unsigned short* kbf = qbf + MD;          // K bf16 / later attn-LN out bf16
    unsigned short* vT  = kbf + MD;          // V bf16, head-transposed [B,H,DK,S]
    unsigned short* hbf = vT  + MD;          // FFN intermediate bf16 (M x F)
    unsigned short* scb = hbf + MF;          // scores / probs bf16 (B*H*S*S)
    unsigned short* wqb = scb + SC;          // bf16 weights (all layers)
    unsigned short* wkb = wqb + LDD;
    unsigned short* wvb = wkb + LDD;
    unsigned short* wob = wvb + LDD;
    unsigned short* w1b = wob + LDD;
    unsigned short* w2b = w1b + LFD;
    unsigned short* ctxbf = qbf;
    unsigned short* abf   = kbf;
    float* yfin = (float*)d_out;

    const dim3 blk(256);

    // one-time weight conversion to bf16
    convert_f32_bf16<<<4096, blk, 0, stream>>>(Wq, wqb, LDD);
    convert_f32_bf16<<<4096, blk, 0, stream>>>(Wk, wkb, LDD);
    convert_f32_bf16<<<4096, blk, 0, stream>>>(Wv, wvb, LDD);
    convert_f32_bf16<<<4096, blk, 0, stream>>>(Wo, wob, LDD);
    convert_f32_bf16<<<4096, blk, 0, stream>>>(W1, w1b, LFD);
    convert_f32_bf16<<<4096, blk, 0, stream>>>(W2, w2b, LFD);

    embed_kernel<<<Mtok, blk, 0, stream>>>(input_ids, token_type, word_emb,
                                           pos_emb, type_emb, x, S, Dd);
    layernorm_rows<<<Mtok, blk, 0, stream>>>(x, emb_g, emb_b, x, xbf, Dd, 1e-12f);

    const dim3 gProj(Dd / 128, Mtok / 128, 1);      // 6 x 32
    const dim3 gScore(S / 128, S / 128, Bb * Hh);   // 4 x 4 x 96
    const dim3 gCtx(1, S / 128, Bb * Hh);           // 1 x 4 x 96
    const dim3 gF1(Ff / 128, Mtok / 128, 1);        // 24 x 32
    const dim3 gF2(Dd / 128, Mtok / 128, 1);        // 6 x 32

    for (int l = 0; l < Ll; ++l) {
        const unsigned short* wqL = wqb + (long long)l * Dd * Dd;
        const unsigned short* wkL = wkb + (long long)l * Dd * Dd;
        const unsigned short* wvL = wvb + (long long)l * Dd * Dd;
        const unsigned short* woL = wob + (long long)l * Dd * Dd;
        const unsigned short* w1L = w1b + (long long)l * Ff * Dd;
        const unsigned short* w2L = w2b + (long long)l * Dd * Ff;
        const float* bqL = bq + (long long)l * Dd;
        const float* bkL = bk + (long long)l * Dd;
        const float* bvL = bv + (long long)l * Dd;
        const float* boL = bo + (long long)l * Dd;
        const float* b1L = b1 + (long long)l * Ff;
        const float* b2L = b2 + (long long)l * Dd;
        const float* agL = ag + (long long)l * Dd;
        const float* abL = ab + (long long)l * Dd;
        const float* fgL = fg + (long long)l * Dd;
        const float* fbL = fb + (long long)l * Dd;

        // Q, K: bf16 out; V: bf16 head-transposed out
        gemm_bf16_wmma<<<gProj, blk, 0, stream>>>(xbf, wqL, bqL, nullptr, nullptr, qbf,
            Mtok, Dd, Dd, Dd, Dd, Dd, 0, 0, 0, 0, 0, 0, 1, 1.f, 1, 0, S, Hh, DK);
        gemm_bf16_wmma<<<gProj, blk, 0, stream>>>(xbf, wkL, bkL, nullptr, nullptr, kbf,
            Mtok, Dd, Dd, Dd, Dd, Dd, 0, 0, 0, 0, 0, 0, 1, 1.f, 1, 0, S, Hh, DK);
        gemm_bf16_wmma<<<gProj, blk, 0, stream>>>(xbf, wvL, bvL, nullptr, nullptr, vT,
            Mtok, Dd, Dd, Dd, Dd, Dd, 0, 0, 0, 0, 0, 0, 1, 1.f, 2, 0, S, Hh, DK);

        // scores = scale * Q K^T with mask -> bf16, batched over (b,h)
        gemm_bf16_wmma<<<gScore, blk, 0, stream>>>(qbf, kbf, nullptr, nullptr, attn_mask, scb,
            S, S, DK, Dd, Dd, S,
            (long long)S * Dd, (long long)DK,
            (long long)S * Dd, (long long)DK,
            (long long)Hh * S * S, (long long)S * S,
            Hh, SCALE, 1, S, S, Hh, DK);

        softmax_rows_bf16<<<Bb * Hh * S, blk, 0, stream>>>(scb, S);

        // ctx = P * V  (V already [N=DK, K=S] per head) -> bf16 [B,S,D]
        gemm_bf16_wmma<<<gCtx, blk, 0, stream>>>(scb, vT, nullptr, nullptr, nullptr, ctxbf,
            S, DK, S, S, S, Dd,
            (long long)Hh * S * S, (long long)S * S,
            (long long)Hh * DK * S, (long long)DK * S,
            (long long)S * Dd, (long long)DK,
            Hh, 1.f, 1, 0, S, Hh, DK);

        // attention out projection + residual(x, f32) -> t (f32), then LN -> abf
        gemm_bf16_wmma<<<gProj, blk, 0, stream>>>(ctxbf, woL, boL, x, nullptr, t,
            Mtok, Dd, Dd, Dd, Dd, Dd, 0, 0, 0, 0, 0, 0, 1, 1.f, 0, 0, S, Hh, DK);
        layernorm_rows<<<Mtok, blk, 0, stream>>>(t, agL, abL, nullptr, abf, Dd, 1e-5f);

        // FFN (no activation; final LN has no residual)
        gemm_bf16_wmma<<<gF1, blk, 0, stream>>>(abf, w1L, b1L, nullptr, nullptr, hbf,
            Mtok, Ff, Dd, Dd, Dd, Ff, 0, 0, 0, 0, 0, 0, 1, 1.f, 1, 0, S, Hh, DK);
        gemm_bf16_wmma<<<gF2, blk, 0, stream>>>(hbf, w2L, b2L, nullptr, nullptr, t,
            Mtok, Dd, Ff, Ff, Ff, Dd, 0, 0, 0, 0, 0, 0, 1, 1.f, 0, 0, S, Hh, DK);

        float* outFp = (l == Ll - 1) ? yfin : x;
        unsigned short* outBp = (l == Ll - 1) ? nullptr : xbf;
        layernorm_rows<<<Mtok, blk, 0, stream>>>(t, fgL, fbL, outFp, outBp, Dd, 1e-5f);
    }
}
